// SummaRunnerSentenceExtractor_1795296329889
// MI455X (gfx1250) — matrix operand order repaired and checked by reading, not compiled, not run on hardware
//
#include <hip/hip_runtime.h>
#include <hip/hip_bf16.h>
#include <math.h>

#define SEQ   128
#define BATCH 256
#define INF_  1024
#define HID   512
#define G3    1536
#define SENTD 256
#define DOCD  512

typedef __attribute__((ext_vector_type(16))) __bf16 v16bf;
typedef __attribute__((ext_vector_type(8)))  float  v8f;

// ---------- scalar helpers ----------
__device__ inline unsigned short f2bf_bits(float f) {
    unsigned u = __float_as_uint(f);
    u += 0x7FFFu + ((u >> 16) & 1u);           // round-to-nearest-even
    return (unsigned short)(u >> 16);
}
__device__ inline float bf2f(unsigned short h) {
    return __uint_as_float(((unsigned)h) << 16);
}
__device__ inline __bf16 usbf(unsigned short v) { return __builtin_bit_cast(__bf16, v); }
__device__ inline float sigmoidf_(float x) { return 1.0f / (1.0f + expf(-x)); }

// ---------- WMMA fragment helpers (CDNA5 16x16x32 bf16, wave32) ----------
__device__ inline v8f wmma_acc(v16bf a, v16bf b, v8f c) {
    return __builtin_amdgcn_wmma_f32_16x16x32_bf16(false, a, false, b, (short)0, c,
                                                   false, false);
}

// A: row-major [M x lda] bf16(ushort). Lane<16: M=m0+lane, K = k0 + {0..7,16..23}
//    Lane>=16: same rows, K = {8..15,24..31} (per ISA 16-bit A table)
__device__ inline v16bf load_a(const unsigned short* A, int lda, int m0, int k0, int lane) {
    int m  = m0 + (lane & 15);
    int kb = k0 + ((lane >> 4) << 3);
    const unsigned short* p = A + (size_t)m * lda + kb;
    v16bf a;
#pragma unroll
    for (int e = 0; e < 8; ++e) { a[e] = usbf(p[e]); a[8 + e] = usbf(p[16 + e]); }
    return a;
}
// A loader from f32 with tanh() applied on the fly (for novelty GEMM)
__device__ inline v16bf load_a_f32_tanh(const float* A, int lda, int m0, int k0, int lane) {
    int m  = m0 + (lane & 15);
    int kb = k0 + ((lane >> 4) << 3);
    const float* p = A + (size_t)m * lda + kb;
    v16bf a;
#pragma unroll
    for (int e = 0; e < 8; ++e) {
        a[e]     = usbf(f2bf_bits(tanhf(p[e])));
        a[8 + e] = usbf(f2bf_bits(tanhf(p[16 + e])));
    }
    return a;
}
// B-matrix = W^T where W is row-major [N x ldw]; lane<16: N=n0+lane, K=k0+{0..7,16..23}
//            lane>=16: K=k0+{8..15,24..31} (same k order as load_a)
__device__ inline v16bf load_b(const unsigned short* W, int ldw, int n0, int k0, int lane) {
    int n  = n0 + (lane & 15);
    int kb = k0 + ((lane >> 4) << 3);
    const unsigned short* p = W + (size_t)n * ldw + kb;
    v16bf b;
#pragma unroll
    for (int e = 0; e < 8; ++e) { b[e] = usbf(p[e]); b[8 + e] = usbf(p[16 + e]); }
    return b;
}

// ---------- conversion ----------
__global__ void k_f32_to_bf16(const float* __restrict__ src, unsigned short* __restrict__ dst, int n) {
    int i = blockIdx.x * blockDim.x + threadIdx.x;
    if (i < n) dst[i] = f2bf_bits(src[i]);
}

// ---------- XP = X @ W_ih^T + b_ih, stored bf16 as [S][B][1536] ----------
// Register-blocked: wave tile 32(M)x64(N), 8 accumulators; block 128x128.
// grid (G3/128=12, 32768/128=256), block 256 (8 waves: 4 in M x 2 in N)
__global__ void k_gemm_xp(const unsigned short* __restrict__ Xbf,
                          const unsigned short* __restrict__ Wbf,
                          const float* __restrict__ b_ih,
                          unsigned short* __restrict__ XP) {
    int wave = threadIdx.x >> 5, lane = threadIdx.x & 31;
    int mw = wave & 3, nw = wave >> 2;
    int m0 = blockIdx.y * 128 + mw * 32;
    int n0 = blockIdx.x * 128 + nw * 64;
    v8f acc[2][4] = {};
    for (int k0 = 0; k0 < INF_; k0 += 32) {
        __builtin_prefetch(Xbf + (size_t)m0 * INF_ + k0 + 32, 0, 1);
        v16bf a0 = load_a(Xbf, INF_, m0,      k0, lane);
        v16bf a1 = load_a(Xbf, INF_, m0 + 16, k0, lane);
        v16bf b0 = load_b(Wbf, INF_, n0,      k0, lane);
        v16bf b1 = load_b(Wbf, INF_, n0 + 16, k0, lane);
        v16bf b2 = load_b(Wbf, INF_, n0 + 32, k0, lane);
        v16bf b3 = load_b(Wbf, INF_, n0 + 48, k0, lane);
        acc[0][0] = wmma_acc(a0, b0, acc[0][0]);
        acc[0][1] = wmma_acc(a0, b1, acc[0][1]);
        acc[0][2] = wmma_acc(a0, b2, acc[0][2]);
        acc[0][3] = wmma_acc(a0, b3, acc[0][3]);
        acc[1][0] = wmma_acc(a1, b0, acc[1][0]);
        acc[1][1] = wmma_acc(a1, b1, acc[1][1]);
        acc[1][2] = wmma_acc(a1, b2, acc[1][2]);
        acc[1][3] = wmma_acc(a1, b3, acc[1][3]);
    }
#pragma unroll
    for (int j = 0; j < 4; ++j) {
        int n = n0 + j * 16 + (lane & 15);
        float bi = b_ih[n];
#pragma unroll
        for (int i = 0; i < 2; ++i) {
#pragma unroll
            for (int r = 0; r < 8; ++r) {
                int m = m0 + i * 16 + r + ((lane >> 4) << 3);   // m = b*SEQ + s
                int s = m & (SEQ - 1);
                int bb = m >> 7;
                XP[((size_t)s * BATCH + bb) * G3 + n] = f2bf_bits(acc[i][j][r] + bi);
            }
        }
    }
}

// ---------- one GRU step: gh = h@W_hh^T + b_hh, fused gates ----------
// Wave tile 32(M)x16(N) with r/z/n accumulators (6 wmma / 5 frag loads).
// grid (HID/16=32, 1), block 256 (8 waves covering all 256 batch rows)
__global__ void k_gru_step(const unsigned short* __restrict__ XP,
                           const unsigned short* __restrict__ Hprev,   // [B][512] bf16
                           const unsigned short* __restrict__ Whh,     // [1536][512] bf16
                           const float* __restrict__ b_hh,
                           unsigned short* __restrict__ Hout,          // [B][512] bf16 slice
                           int s) {
    int wave = threadIdx.x >> 5, lane = threadIdx.x & 31;
    int m0 = wave * 32;
    int n0 = blockIdx.x * 16;
    v8f ar[2] = {}, az[2] = {}, an[2] = {};
    for (int k0 = 0; k0 < HID; k0 += 32) {
        v16bf a0 = load_a(Hprev, HID, m0,      k0, lane);
        v16bf a1 = load_a(Hprev, HID, m0 + 16, k0, lane);
        v16bf br = load_b(Whh, HID, n0,           k0, lane);
        v16bf bz = load_b(Whh, HID, n0 + HID,     k0, lane);
        v16bf bn = load_b(Whh, HID, n0 + 2 * HID, k0, lane);
        ar[0] = wmma_acc(a0, br, ar[0]);
        az[0] = wmma_acc(a0, bz, az[0]);
        an[0] = wmma_acc(a0, bn, an[0]);
        ar[1] = wmma_acc(a1, br, ar[1]);
        az[1] = wmma_acc(a1, bz, az[1]);
        an[1] = wmma_acc(a1, bn, an[1]);
    }
    int n = n0 + (lane & 15);
    float bhr = b_hh[n], bhz = b_hh[n + HID], bhn = b_hh[n + 2 * HID];
#pragma unroll
    for (int i = 0; i < 2; ++i) {
#pragma unroll
        for (int r = 0; r < 8; ++r) {
            int bb = m0 + i * 16 + r + ((lane >> 4) << 3);
            size_t xbase = ((size_t)s * BATCH + bb) * G3 + n;
            float xr = bf2f(XP[xbase]);
            float xz = bf2f(XP[xbase + HID]);
            float xn = bf2f(XP[xbase + 2 * HID]);
            float rg = sigmoidf_(xr + ar[i][r] + bhr);
            float zg = sigmoidf_(xz + az[i][r] + bhz);
            float ng = tanhf(xn + rg * (an[i][r] + bhn));
            float hp = bf2f(Hprev[(size_t)bb * HID + n]);
            float hnew = (1.0f - zg) * ng + zg * hp;
            Hout[(size_t)bb * HID + n] = f2bf_bits(hnew);
        }
    }
}

// ---------- sent = relu([fwd|bwd] @ W_sent^T + b), rows r = s*B+b ----------
// Register-blocked like k_gemm_xp. grid (SENTD/128=2, 32768/128=256)
__global__ void k_gemm_sent(const unsigned short* __restrict__ Hf,
                            const unsigned short* __restrict__ Hb,
                            const unsigned short* __restrict__ Wsent,   // [256][1024]
                            const float* __restrict__ b_sent,
                            float* __restrict__ sent_f) {
    int wave = threadIdx.x >> 5, lane = threadIdx.x & 31;
    int mw = wave & 3, nw = wave >> 2;
    int m0 = blockIdx.y * 128 + mw * 32;
    int n0 = blockIdx.x * 128 + nw * 64;
    v8f acc[2][4] = {};
    for (int k0 = 0; k0 < 2 * HID; k0 += 32) {
        const unsigned short* A = (k0 < HID) ? Hf : Hb;
        int kk = k0 & (HID - 1);
        __builtin_prefetch(A + (size_t)m0 * HID + kk + 32, 0, 1);
        v16bf a0 = load_a(A, HID, m0,      kk, lane);
        v16bf a1 = load_a(A, HID, m0 + 16, kk, lane);
        v16bf b0 = load_b(Wsent, 2 * HID, n0,      k0, lane);
        v16bf b1 = load_b(Wsent, 2 * HID, n0 + 16, k0, lane);
        v16bf b2 = load_b(Wsent, 2 * HID, n0 + 32, k0, lane);
        v16bf b3 = load_b(Wsent, 2 * HID, n0 + 48, k0, lane);
        acc[0][0] = wmma_acc(a0, b0, acc[0][0]);
        acc[0][1] = wmma_acc(a0, b1, acc[0][1]);
        acc[0][2] = wmma_acc(a0, b2, acc[0][2]);
        acc[0][3] = wmma_acc(a0, b3, acc[0][3]);
        acc[1][0] = wmma_acc(a1, b0, acc[1][0]);
        acc[1][1] = wmma_acc(a1, b1, acc[1][1]);
        acc[1][2] = wmma_acc(a1, b2, acc[1][2]);
        acc[1][3] = wmma_acc(a1, b3, acc[1][3]);
    }
#pragma unroll
    for (int j = 0; j < 4; ++j) {
        int n = n0 + j * 16 + (lane & 15);
        float bs = b_sent[n];
#pragma unroll
        for (int i = 0; i < 2; ++i) {
#pragma unroll
            for (int r = 0; r < 8; ++r) {
                int m = m0 + i * 16 + r + ((lane >> 4) << 3);
                float v = acc[i][j][r] + bs;
                sent_f[(size_t)m * SENTD + n] = v > 0.0f ? v : 0.0f;
            }
        }
    }
}

// ---------- avg over S (-> bf16) and content logits ----------
__global__ void k_avg_content(const float* __restrict__ sent_f,
                              const float* __restrict__ Wc, const float* __restrict__ bc,
                              const int* __restrict__ num,
                              unsigned short* __restrict__ avg_bf,
                              float* __restrict__ stat) {
    int b = blockIdx.x, n = threadIdx.x;
    __shared__ float red[256];
    float wc = Wc[n];
    float accum = 0.0f;
    for (int s = 0; s < SEQ; ++s) {
        float v = sent_f[((size_t)s * BATCH + b) * SENTD + n];
        accum += v;
        red[n] = v * wc;
        __syncthreads();
        for (int off = 128; off > 0; off >>= 1) {
            if (n < off) red[n] += red[n + off];
            __syncthreads();
        }
        if (n == 0) stat[s * BATCH + b] = red[0] + bc[0];
        __syncthreads();
    }
    avg_bf[(size_t)b * SENTD + n] = f2bf_bits(accum / (float)num[b]);
}

// ---------- doc MLP: t1 = tanh(avg@W1^T+b1) ; doc = t1@W2^T+b2 ----------
__global__ void k_gemm_doc1(const unsigned short* __restrict__ avg_bf,
                            const unsigned short* __restrict__ W1,  // [512][256]
                            const float* __restrict__ b1,
                            unsigned short* __restrict__ t1_bf) {
    int wave = threadIdx.x >> 5, lane = threadIdx.x & 31;
    int n0 = blockIdx.x * 16;
    int m0 = blockIdx.y * 128 + wave * 16;
    v8f acc = {};
    for (int k0 = 0; k0 < SENTD; k0 += 32) {
        v16bf a = load_a(avg_bf, SENTD, m0, k0, lane);
        v16bf b = load_b(W1, SENTD, n0, k0, lane);
        acc = wmma_acc(a, b, acc);
    }
    int n = n0 + (lane & 15);
    float bb1 = b1[n];
#pragma unroll
    for (int r = 0; r < 8; ++r) {
        int m = m0 + r + ((lane >> 4) << 3);
        t1_bf[(size_t)m * DOCD + n] = f2bf_bits(tanhf(acc[r] + bb1));
    }
}
__global__ void k_gemm_doc2(const unsigned short* __restrict__ t1_bf,
                            const unsigned short* __restrict__ W2,  // [256][512]
                            const float* __restrict__ b2,
                            float* __restrict__ doc_f) {
    int wave = threadIdx.x >> 5, lane = threadIdx.x & 31;
    int n0 = blockIdx.x * 16;
    int m0 = blockIdx.y * 128 + wave * 16;
    v8f acc = {};
    for (int k0 = 0; k0 < DOCD; k0 += 32) {
        v16bf a = load_a(t1_bf, DOCD, m0, k0, lane);
        v16bf b = load_b(W2, DOCD, n0, k0, lane);
        acc = wmma_acc(a, b, acc);
    }
    int n = n0 + (lane & 15);
    float bb2 = b2[n];
#pragma unroll
    for (int r = 0; r < 8; ++r) {
        int m = m0 + r + ((lane >> 4) << 3);
        doc_f[(size_t)m * SENTD + n] = acc[r] + bb2;
    }
}

// ---------- static = content + salience + pos + seg + bias ----------
__global__ void k_static(const float* __restrict__ sent_f, const float* __restrict__ doc_f,
                         float* __restrict__ stat, const int* __restrict__ num,
                         const float* __restrict__ pos_emb, const float* __restrict__ W_pos,
                         const float* __restrict__ seg_emb, const float* __restrict__ W_seg,
                         const float* __restrict__ bias) {
    int s = blockIdx.x, b = threadIdx.x;
    size_t srow = ((size_t)s * BATCH + b) * SENTD;
    size_t drow = (size_t)b * SENTD;
    float sal = 0.0f;
    for (int n = 0; n < SENTD; ++n) sal += sent_f[srow + n] * doc_f[drow + n];
    int ap = s + 1;
    int apc = ap > 25 ? 25 : ap;
    float pl = 0.0f;
    for (int j = 0; j < 50; ++j) pl += pos_emb[apc * 50 + j] * W_pos[j];
    float chunk = rintf((float)num[b] * 0.25f);   // round-half-even like jnp.round
    float rp = ceilf((float)ap / chunk);
    rp = fminf(fmaxf(rp, 0.0f), 4.0f);
    int ri = (int)rp;
    float sl = 0.0f;
    for (int j = 0; j < 50; ++j) sl += seg_emb[ri * 50 + j] * W_seg[j];
    stat[s * BATCH + b] += sal + pl + sl + bias[0];
}

// ---------- novelty scan: u = tanh(summary) @ W_sim^T (WMMA, tanh fused) ----------
__global__ void k_gemm_nov(const float* __restrict__ summary,
                           const unsigned short* __restrict__ Wsim,   // [256][256]
                           float* __restrict__ u) {
    int wave = threadIdx.x >> 5, lane = threadIdx.x & 31;
    int n0 = blockIdx.x * 16;
    int m0 = blockIdx.y * 128 + wave * 16;
    v8f acc = {};
    for (int k0 = 0; k0 < SENTD; k0 += 32) {
        v16bf a = load_a_f32_tanh(summary, SENTD, m0, k0, lane);
        v16bf b = load_b(Wsim, SENTD, n0, k0, lane);
        acc = wmma_acc(a, b, acc);
    }
    int n = n0 + (lane & 15);
#pragma unroll
    for (int r = 0; r < 8; ++r) {
        int m = m0 + r + ((lane >> 4) << 3);
        u[(size_t)m * SENTD + n] = acc[r];
    }
}
__global__ void k_nov_update(const float* __restrict__ sent_f, const float* __restrict__ u,
                             const float* __restrict__ stat, float* __restrict__ summary,
                             float* __restrict__ out, int s) {
    int b = blockIdx.x, i = threadIdx.x;
    __shared__ float red[256];
    size_t srow = ((size_t)s * BATCH + b) * SENTD;
    float sv = sent_f[srow + i];
    red[i] = sv * u[(size_t)b * SENTD + i];
    __syncthreads();
    for (int off = 128; off > 0; off >>= 1) {
        if (i < off) red[i] += red[i + off];
        __syncthreads();
    }
    float logit = stat[s * BATCH + b] - red[0];
    float sg = sigmoidf_(logit);
    summary[(size_t)b * SENTD + i] += sv * sg;
    if (i == 0) out[(size_t)b * SEQ + s] = logit;
}

// ============================ host side ============================
extern "C" void kernel_launch(void* const* d_in, const int* in_sizes, int n_in,
                              void* d_out, int out_size, void* d_ws, size_t ws_size,
                              hipStream_t stream) {
    const float* X        = (const float*)d_in[0];
    const int*   num      = (const int*)d_in[1];
    const float* W_ih_f   = (const float*)d_in[2];
    const float* W_hh_f   = (const float*)d_in[3];
    const float* b_ih_f   = (const float*)d_in[4];
    const float* b_hh_f   = (const float*)d_in[5];
    const float* W_ih_b   = (const float*)d_in[6];
    const float* W_hh_b   = (const float*)d_in[7];
    const float* b_ih_b   = (const float*)d_in[8];
    const float* b_hh_b   = (const float*)d_in[9];
    const float* W_sent   = (const float*)d_in[10];
    const float* b_sent   = (const float*)d_in[11];
    const float* W_cont   = (const float*)d_in[12];
    const float* b_cont   = (const float*)d_in[13];
    const float* W_doc1   = (const float*)d_in[14];
    const float* b_doc1   = (const float*)d_in[15];
    const float* W_doc2   = (const float*)d_in[16];
    const float* b_doc2   = (const float*)d_in[17];
    const float* W_sim    = (const float*)d_in[18];
    const float* bias     = (const float*)d_in[19];
    const float* pos_emb  = (const float*)d_in[20];
    const float* W_pos    = (const float*)d_in[21];
    const float* seg_emb  = (const float*)d_in[22];
    const float* W_seg    = (const float*)d_in[23];
    (void)in_sizes; (void)n_in; (void)out_size; (void)ws_size;

    char* ws = (char*)d_ws;
    size_t off = 0;
    auto alloc = [&](size_t bytes) -> char* {
        off = (off + 255) & ~(size_t)255;
        char* p = ws + off;
        off += bytes;
        return p;
    };
    const size_t MROWS = (size_t)BATCH * SEQ;                       // 32768
    unsigned short* Xbf   = (unsigned short*)alloc(MROWS * INF_ * 2);
    unsigned short* Wihf  = (unsigned short*)alloc((size_t)G3 * INF_ * 2);
    unsigned short* Wihb  = (unsigned short*)alloc((size_t)G3 * INF_ * 2);
    unsigned short* Whhf  = (unsigned short*)alloc((size_t)G3 * HID * 2);
    unsigned short* Whhb  = (unsigned short*)alloc((size_t)G3 * HID * 2);
    unsigned short* Wsnt  = (unsigned short*)alloc((size_t)SENTD * 2 * HID * 2);
    unsigned short* Wsm   = (unsigned short*)alloc((size_t)SENTD * SENTD * 2);
    unsigned short* Wd1   = (unsigned short*)alloc((size_t)DOCD * SENTD * 2);
    unsigned short* Wd2   = (unsigned short*)alloc((size_t)SENTD * DOCD * 2);
    unsigned short* XPf   = (unsigned short*)alloc((size_t)SEQ * BATCH * G3 * 2);
    unsigned short* XPb   = (unsigned short*)alloc((size_t)SEQ * BATCH * G3 * 2);
    unsigned short* Hf    = (unsigned short*)alloc((size_t)SEQ * BATCH * HID * 2);
    unsigned short* Hb    = (unsigned short*)alloc((size_t)SEQ * BATCH * HID * 2);
    unsigned short* hzero = (unsigned short*)alloc((size_t)BATCH * HID * 2);
    float*          sentf = (float*)alloc(MROWS * SENTD * 4);
    unsigned short* avgbf = (unsigned short*)alloc((size_t)BATCH * SENTD * 2);
    unsigned short* t1bf  = (unsigned short*)alloc((size_t)BATCH * DOCD * 2);
    float*          docf  = (float*)alloc((size_t)BATCH * SENTD * 4);
    float*          statf = (float*)alloc((size_t)SEQ * BATCH * 4);
    float*          ubuf  = (float*)alloc((size_t)BATCH * SENTD * 4);
    float*          summ  = (float*)alloc((size_t)BATCH * SENTD * 4);

    hipMemsetAsync(hzero, 0, (size_t)BATCH * HID * 2, stream);
    hipMemsetAsync(summ, 0, (size_t)BATCH * SENTD * 4, stream);

    auto conv = [&](const float* s, unsigned short* d, size_t n) {
        k_f32_to_bf16<<<dim3((unsigned)((n + 255) / 256)), dim3(256), 0, stream>>>(s, d, (int)n);
    };
    conv(X,      Xbf,  MROWS * INF_);
    conv(W_ih_f, Wihf, (size_t)G3 * INF_);
    conv(W_ih_b, Wihb, (size_t)G3 * INF_);
    conv(W_hh_f, Whhf, (size_t)G3 * HID);
    conv(W_hh_b, Whhb, (size_t)G3 * HID);
    conv(W_sent, Wsnt, (size_t)SENTD * 2 * HID);
    conv(W_sim,  Wsm,  (size_t)SENTD * SENTD);
    conv(W_doc1, Wd1,  (size_t)DOCD * SENTD);
    conv(W_doc2, Wd2,  (size_t)SENTD * DOCD);

    // x-projection GEMMs (both directions), 128x128 block tiles
    k_gemm_xp<<<dim3(G3 / 128, (unsigned)(MROWS / 128)), 256, 0, stream>>>(Xbf, Wihf, b_ih_f, XPf);
    k_gemm_xp<<<dim3(G3 / 128, (unsigned)(MROWS / 128)), 256, 0, stream>>>(Xbf, Wihb, b_ih_b, XPb);

    // forward GRU scan
    for (int t = 0; t < SEQ; ++t) {
        const unsigned short* hp = (t == 0) ? hzero : Hf + (size_t)(t - 1) * BATCH * HID;
        k_gru_step<<<dim3(HID / 16, 1), 256, 0, stream>>>(
            XPf, hp, Whhf, b_hh_f, Hf + (size_t)t * BATCH * HID, t);
    }
    // backward GRU scan
    for (int t = 0; t < SEQ; ++t) {
        int s = SEQ - 1 - t;
        const unsigned short* hp = (t == 0) ? hzero : Hb + (size_t)(s + 1) * BATCH * HID;
        k_gru_step<<<dim3(HID / 16, 1), 256, 0, stream>>>(
            XPb, hp, Whhb, b_hh_b, Hb + (size_t)s * BATCH * HID, s);
    }

    // sent projection
    k_gemm_sent<<<dim3(SENTD / 128, (unsigned)(MROWS / 128)), 256, 0, stream>>>(
        Hf, Hb, Wsnt, b_sent, sentf);

    // avg + content
    k_avg_content<<<dim3(BATCH), 256, 0, stream>>>(sentf, W_cont, b_cont, num, avgbf, statf);

    // doc MLP
    k_gemm_doc1<<<dim3(DOCD / 16, BATCH / 128), 256, 0, stream>>>(avgbf, Wd1, b_doc1, t1bf);
    k_gemm_doc2<<<dim3(SENTD / 16, BATCH / 128), 256, 0, stream>>>(t1bf, Wd2, b_doc2, docf);

    // static logits
    k_static<<<dim3(SEQ), 256, 0, stream>>>(sentf, docf, statf, num,
                                            pos_emb, W_pos, seg_emb, W_seg, bias);

    // novelty scan
    for (int s = 0; s < SEQ; ++s) {
        k_gemm_nov<<<dim3(SENTD / 16, BATCH / 128), 256, 0, stream>>>(summ, Wsm, ubuf);
        k_nov_update<<<dim3(BATCH), 256, 0, stream>>>(sentf, ubuf, statf, summ,
                                                      (float*)d_out, s);
    }
}
